// NeuralCDE_67997922230775
// MI455X (gfx1250) — matrix operand order repaired and checked by reading, not compiled, not run on hardware
//
#include <hip/hip_runtime.h>
#include <hip/hip_bf16.h>
#include <math.h>

// ---------------- problem constants (from reference setup) ----------------
#define T_LEN   128
#define NI      127          // intervals
#define B_TOT   64
#define OBS     32
#define C_DIM   33           // obs + 1
#define HID     64
#define WID     128
#define FOUT    2112         // HID * C_DIM
#define OUTD    32
#define NSUB    4

typedef __attribute__((ext_vector_type(16))) _Float16 v16h;
typedef __attribute__((ext_vector_type(8)))  _Float16 v8h;
typedef __attribute__((ext_vector_type(8)))  float    v8f;

#define GLOBAL_AS __attribute__((address_space(1)))
#define LDS_AS    __attribute__((address_space(3)))

#if defined(__has_builtin)
#  if __has_builtin(__builtin_amdgcn_global_load_async_to_lds_b32)
#    define HAVE_ASYNC_LDS 1
#  endif
#endif

// ---------------- fast activations ----------------------------------------
__device__ __forceinline__ float fast_softplus(float x) {
    return __logf(1.f + __expf(x));          // v_exp_f32 + v_log_f32
}
__device__ __forceinline__ float fast_tanh(float x) {
#if defined(__has_builtin) && __has_builtin(__builtin_amdgcn_tanhf)
    return __builtin_amdgcn_tanhf(x);        // v_tanh_f32 (gfx1250 TRANS op)
#else
    float e = __expf(2.f * x);
    return 1.f - 2.f * __builtin_amdgcn_rcpf(e + 1.f);
#endif
}

// ---------------- WMMA operand loaders (CDNA5 16x16x32 f16 layouts) -------
// A (16x32 f16): lanes 0-15 row M=lane, halves0-7=K 0..7, halves8-15=K 16..23
//                lanes16-31 row M=lane-16, halves0-7=K 8..15, halves8-15=K 24..31
__device__ __forceinline__ v16h loadA_g(const _Float16* W, int ldk, int mt,
                                        int kc, int lane) {
    int m  = mt * 16 + (lane & 15);
    int kb = kc * 32 + ((lane & 16) ? 8 : 0);
    const _Float16* p = W + (size_t)m * ldk + kb;
    union { v16h v; v8h h[2]; } u;
    u.h[0] = *(const v8h*)(p);        // K kb..kb+7
    u.h[1] = *(const v8h*)(p + 16);   // K kb+16..kb+23
    return u.v;
}
// B (32x16 f16): N = lane&15; lanes 0-15 hold K 0..7,16..23, lanes 16-31 hold K 8..15,24..31
// act is LDS, row-major [n][ldk] (f16) -> two 16B runs per lane (p, p+16)
__device__ __forceinline__ v16h loadB_lds(const _Float16* act, int ldk,
                                          int kc, int lane) {
    int n  = lane & 15;
    int kb = kc * 32 + ((lane & 16) ? 8 : 0);
    const _Float16* p = act + n * ldk + kb;
    union { v16h v; v8h h[2]; } u;
    u.h[0] = *(const v8h*)(p);
    u.h[1] = *(const v8h*)(p + 16);
    return u.v;
}

// One softplus layer: waves 0..7 each own one 16-row M tile (M=128).
// Operands are fully preloaded so the loads clause and overlap; WMMAs chain on C.
template <int K>
__device__ __forceinline__ void layer_softplus(const _Float16* __restrict__ W,
                                               const float* __restrict__ bias,
                                               const _Float16* __restrict__ src,
                                               _Float16* __restrict__ dst,
                                               int w, int lane) {
    if (w < 8) {
        constexpr int NK = K / 32;
        v16h a[NK], b[NK];
#pragma unroll
        for (int kc = 0; kc < NK; ++kc) {
            a[kc] = loadA_g(W, K, w, kc, lane);
            b[kc] = loadB_lds(src, WID, kc, lane);
        }
        v8f acc = {};
#pragma unroll
        for (int kc = 0; kc < NK; ++kc)
            acc = __builtin_amdgcn_wmma_f32_16x16x32_f16(
                false, a[kc], false, b[kc], (short)0, acc, false, false);
        const int n     = lane & 15;
        const int off   = (lane & 16) ? 8 : 0;
        const int mbase = w * 16 + off;
        const float4 b0 = *(const float4*)(bias + mbase);
        const float4 b1 = *(const float4*)(bias + mbase + 4);
        const float bb[8] = {b0.x, b0.y, b0.z, b0.w, b1.x, b1.y, b1.z, b1.w};
        v8h st;
#pragma unroll
        for (int j = 0; j < 8; ++j)
            st[j] = (_Float16)fast_softplus(acc[j] + bb[j]);
        *(v8h*)(dst + n * WID + mbase) = st;     // ds_store_b128
    }
}

// ---------------- prologue kernels ----------------------------------------
__global__ void cvt_f32_to_f16(const float* __restrict__ src,
                               _Float16* __restrict__ dst, int n) {
    int i = blockIdx.x * blockDim.x + threadIdx.x;
    if (i < n) dst[i] = (_Float16)src[i];
}

// Hermite coefficients d0, c, b per (batch, interval, channel)
__global__ void coef_kernel(const float* __restrict__ ts,
                            const float* __restrict__ ys,
                            float* __restrict__ d0g, float* __restrict__ cg,
                            float* __restrict__ bg) {
    int idx = blockIdx.x * blockDim.x + threadIdx.x;
    if (idx >= B_TOT * NI * C_DIM) return;
    int cc  = idx % C_DIM;
    int rem = idx / C_DIM;
    int i   = rem % NI;
    int bb  = rem / NI;
    float ti = ts[i], tip = ts[i + 1];
    float dt = tip - ti;
    float yi  = (cc == 0) ? ti  : ys[((size_t)bb * T_LEN + i) * OBS + cc - 1];
    float yip = (cc == 0) ? tip : ys[((size_t)bb * T_LEN + i + 1) * OBS + cc - 1];
    float di = (yip - yi) / dt;
    float dprev;
    if (i == 0) {
        dprev = di;
    } else {
        float tim = ts[i - 1];
        float yim = (cc == 0) ? tim : ys[((size_t)bb * T_LEN + i - 1) * OBS + cc - 1];
        dprev = (yi - yim) / (ti - tim);
    }
    float d0 = dprev;                               // deriv[:-1]
    float d1 = di;                                  // deriv[1:]
    d0g[idx] = d0;
    cg[idx]  = (3.f * di - 2.f * d0 - d1) / dt;
    bg[idx]  = (d0 + d1 - 2.f * di) / (dt * dt);
}

// Initial ReLU MLP -> y0, plus output row t=0.  One block per batch element.
__global__ void init_kernel(const float* __restrict__ ts, const float* __restrict__ ys,
                            const float* __restrict__ iW0, const float* __restrict__ ib0,
                            const float* __restrict__ iWh, const float* __restrict__ ibh,
                            const float* __restrict__ iWo, const float* __restrict__ ibo,
                            const float* __restrict__ lW,  const float* __restrict__ lb,
                            float* __restrict__ y0g, float* __restrict__ out) {
    __shared__ float tys0[C_DIM];
    __shared__ float h0[WID];
    __shared__ float h1[WID];
    __shared__ float y0s[HID];
    const int b = blockIdx.x;
    const int r = threadIdx.x;  // 128 threads
    if (r < C_DIM)
        tys0[r] = (r == 0) ? ts[0] : ys[((size_t)b * T_LEN) * OBS + r - 1];
    __syncthreads();
    float acc = ib0[r];
    for (int c = 0; c < C_DIM; ++c) acc += iW0[r * C_DIM + c] * tys0[c];
    h0[r] = fmaxf(acc, 0.f);
    __syncthreads();
    for (int k = 0; k < 3; ++k) {
        float a2 = ibh[k * WID + r];
        for (int c = 0; c < WID; ++c) a2 += iWh[((size_t)k * WID + r) * WID + c] * h0[c];
        h1[r] = fmaxf(a2, 0.f);
        __syncthreads();
        h0[r] = h1[r];
        __syncthreads();
    }
    if (r < HID) {
        float a3 = ibo[r];
        for (int c = 0; c < WID; ++c) a3 += iWo[r * WID + c] * h0[c];
        y0s[r] = a3;
        y0g[(size_t)b * HID + r] = a3;
    }
    __syncthreads();
    if (r < OUTD) {
        float a4 = lb[r];
        for (int c = 0; c < HID; ++c) a4 += lW[r * HID + c] * y0s[c];
        out[((size_t)b * T_LEN) * OUTD + r] = a4;
    }
}

// ---------------- main sequential CDE integrator --------------------------
// grid = 4 blocks (16 batch elements each), block = 1024 threads = 32 waves.
__global__ void __launch_bounds__(1024)
ode_kernel(const float* __restrict__ ts,
           const _Float16* __restrict__ fW0h, const float* __restrict__ fb0,
           const _Float16* __restrict__ fWhh, const float* __restrict__ fbh,
           const _Float16* __restrict__ fWoh, const float* __restrict__ fbo,
           const float* __restrict__ lW, const float* __restrict__ lb,
           const float* __restrict__ y0g,
           const float* __restrict__ d0g, const float* __restrict__ cg,
           const float* __restrict__ bg, float* __restrict__ out) {
    __shared__ _Float16 actA[16 * WID];     // activations [n][k] (f16)
    __shared__ _Float16 actB[16 * WID];
    __shared__ float derivS[HID * 16];      // vf output (accumulated) [h][n]
    __shared__ float xdotS[C_DIM * 16];     // path derivative [c][n]
    __shared__ float ysnap[HID * 16];       // y copy for output projection
    __shared__ float coefStage[3 * C_DIM * 16];  // interval coeffs staged via async

    const int tid  = threadIdx.x;
    const int w    = tid >> 5;
    const int lane = tid & 31;
    const int n0   = blockIdx.x * 16;       // batch slice start
    const int hOwn = tid >> 4;              // owned state element: y[hOwn][nOwn]
    const int nOwn = tid & 15;

    float yreg = y0g[(size_t)(n0 + nOwn) * HID + hOwn];

    for (int t = 0; t < NI; ++t) {
        // ---- stage this interval's Hermite coefficients into LDS ---------
        if (tid < C_DIM * 16) {
            int ci = tid >> 4, nn = tid & 15;
            size_t g = ((size_t)(n0 + nn) * NI + t) * C_DIM + ci;
#ifdef HAVE_ASYNC_LDS
            __builtin_amdgcn_global_load_async_to_lds_b32(
                (GLOBAL_AS int*)(d0g + g),
                (LDS_AS int*)&coefStage[tid], 0, 0);
            __builtin_amdgcn_global_load_async_to_lds_b32(
                (GLOBAL_AS int*)(cg + g),
                (LDS_AS int*)&coefStage[C_DIM * 16 + tid], 0, 0);
            __builtin_amdgcn_global_load_async_to_lds_b32(
                (GLOBAL_AS int*)(bg + g),
                (LDS_AS int*)&coefStage[2 * C_DIM * 16 + tid], 0, 0);
#else
            coefStage[tid]                  = d0g[g];
            coefStage[C_DIM * 16 + tid]     = cg[g];
            coefStage[2 * C_DIM * 16 + tid] = bg[g];
#endif
        }
        const float dt = ts[t + 1] - ts[t];
        const float hs = 0.25f * dt;
#ifdef HAVE_ASYNC_LDS
#  if defined(__has_builtin) && __has_builtin(__builtin_amdgcn_s_wait_asynccnt)
        __builtin_amdgcn_s_wait_asynccnt(0);
#  else
        asm volatile("s_wait_asynccnt 0x0" ::: "memory");
#  endif
#endif
        // each staging thread consumes only its own slots -> registers
        float d0v = 0.f, cv = 0.f, bv = 0.f;
        if (tid < C_DIM * 16) {
            d0v = coefStage[tid];
            cv  = coefStage[C_DIM * 16 + tid];
            bv  = coefStage[2 * C_DIM * 16 + tid];
        }

        for (int sub = 0; sub < NSUB; ++sub) {
            const float s0 = sub * hs;
            float kv = 0.f, kacc = 0.f;
            for (int st = 0; st < 4; ++st) {
                const float aIn  = (st == 0) ? 0.f : (st == 3) ? hs : 0.5f * hs;
                const float sloc = s0 + aIn;
                // stage input y + a*k_prev -> actA (f16), clear accumulators
                actA[nOwn * WID + hOwn] = (_Float16)(yreg + aIn * kv);
                derivS[tid] = 0.f;
                if (tid < C_DIM * 16)
                    xdotS[tid] = d0v + (2.f * cv + 3.f * bv * sloc) * sloc;
                __syncthreads();
                // MLP trunk: 128x64, then 3x 128x128, all softplus
                layer_softplus<HID>(fW0h,                 fb0,           actA, actB, w, lane);
                __syncthreads();
                layer_softplus<WID>(fWhh,                 fbh,           actB, actA, w, lane);
                __syncthreads();
                layer_softplus<WID>(fWhh + WID * WID,     fbh + WID,     actA, actB, w, lane);
                __syncthreads();
                layer_softplus<WID>(fWhh + 2 * WID * WID, fbh + 2 * WID, actB, actA, w, lane);
                __syncthreads();
                // big layer: 2112x128, tanh, fused contraction with xdot
                for (int mt = w; mt < FOUT / 16; mt += 32) {
                    if (mt + 32 < FOUT / 16)
                        __builtin_prefetch(fWoh + (size_t)(mt + 32) * 16 * WID, 0, 3);
                    v16h a[4], bm[4];
#pragma unroll
                    for (int kc = 0; kc < 4; ++kc) {
                        a[kc]  = loadA_g(fWoh, WID, mt, kc, lane);
                        bm[kc] = loadB_lds(actA, WID, kc, lane);
                    }
                    v8f acc = {};
#pragma unroll
                    for (int kc = 0; kc < 4; ++kc)
                        acc = __builtin_amdgcn_wmma_f32_16x16x32_f16(
                            false, a[kc], false, bm[kc], (short)0, acc, false, false);
                    const int n   = lane & 15;
                    const int off = (lane & 16) ? 8 : 0;
                    const int rb  = mt * 16 + off;
                    const float4 fb4a = *(const float4*)(fbo + rb);
                    const float4 fb4b = *(const float4*)(fbo + rb + 4);
                    const float bb[8] = {fb4a.x, fb4a.y, fb4a.z, fb4a.w,
                                         fb4b.x, fb4b.y, fb4b.z, fb4b.w};
#pragma unroll
                    for (int j = 0; j < 8; ++j) {
                        int row  = rb + j;
                        float g  = fast_tanh(acc[j] + bb[j]);
                        int hidx = row / C_DIM;
                        int cidx = row - hidx * C_DIM;
                        atomicAdd(&derivS[hidx * 16 + n], g * xdotS[cidx * 16 + n]);
                    }
                }
                __syncthreads();
                kv = derivS[tid];  // k_{st+1} for owned element
                kacc += ((st == 0 || st == 3) ? hs * (1.f / 6.f) : hs * (1.f / 3.f)) * kv;
            }
            yreg += kacc;          // RK4 update
        }
        // output projection for this interval end: out[n][t+1][:] = lW @ y + lb
        ysnap[tid] = yreg;         // tid == hOwn*16 + nOwn
        __syncthreads();
        if (tid < 16 * OUTD) {
            int nn = tid >> 5, o = tid & 31;
            float a = lb[o];
#pragma unroll 8
            for (int h = 0; h < HID; ++h) a += ysnap[h * 16 + nn] * lW[o * HID + h];
            out[((size_t)(n0 + nn) * T_LEN + (t + 1)) * OUTD + o] = a;
        }
        __syncthreads();
    }
}

// ---------------- host launch ---------------------------------------------
extern "C" void kernel_launch(void* const* d_in, const int* in_sizes, int n_in,
                              void* d_out, int out_size, void* d_ws, size_t ws_size,
                              hipStream_t stream) {
    (void)in_sizes; (void)n_in; (void)out_size; (void)ws_size;
    const float* ts  = (const float*)d_in[0];
    const float* ys  = (const float*)d_in[1];
    const float* iW0 = (const float*)d_in[2];
    const float* ib0 = (const float*)d_in[3];
    const float* iWh = (const float*)d_in[4];
    const float* ibh = (const float*)d_in[5];
    const float* iWo = (const float*)d_in[6];
    const float* ibo = (const float*)d_in[7];
    const float* fW0 = (const float*)d_in[8];
    const float* fb0 = (const float*)d_in[9];
    const float* fWh = (const float*)d_in[10];
    const float* fbh = (const float*)d_in[11];
    const float* fWo = (const float*)d_in[12];
    const float* fbo = (const float*)d_in[13];
    const float* lW  = (const float*)d_in[14];
    const float* lb  = (const float*)d_in[15];
    float* out = (float*)d_out;

    // workspace carve-up (256B aligned)
    char*  ws = (char*)d_ws;
    size_t o  = 0;
    auto take = [&](size_t bytes) -> char* {
        char* p = ws + o;
        o += (bytes + 255) & ~(size_t)255;
        return p;
    };
    const int nW0 = WID * HID;           // 8192
    const int nWh = 3 * WID * WID;       // 49152
    const int nWo = FOUT * WID;          // 270336
    const int nCo = B_TOT * NI * C_DIM;  // 268224
    _Float16* fW0h = (_Float16*)take((size_t)nW0 * 2);
    _Float16* fWhh = (_Float16*)take((size_t)nWh * 2);
    _Float16* fWoh = (_Float16*)take((size_t)nWo * 2);
    float* d0g = (float*)take((size_t)nCo * 4);
    float* cg  = (float*)take((size_t)nCo * 4);
    float* bg  = (float*)take((size_t)nCo * 4);
    float* y0g = (float*)take((size_t)B_TOT * HID * 4);

    cvt_f32_to_f16<<<(nW0 + 255) / 256, 256, 0, stream>>>(fW0, fW0h, nW0);
    cvt_f32_to_f16<<<(nWh + 255) / 256, 256, 0, stream>>>(fWh, fWhh, nWh);
    cvt_f32_to_f16<<<(nWo + 255) / 256, 256, 0, stream>>>(fWo, fWoh, nWo);
    coef_kernel<<<(nCo + 255) / 256, 256, 0, stream>>>(ts, ys, d0g, cg, bg);
    init_kernel<<<B_TOT, WID, 0, stream>>>(ts, ys, iW0, ib0, iWh, ibh,
                                           iWo, ibo, lW, lb, y0g, out);
    ode_kernel<<<B_TOT / 16, 1024, 0, stream>>>(ts, fW0h, fb0, fWhh, fbh,
                                                fWoh, fbo, lW, lb, y0g,
                                                d0g, cg, bg, out);
}
